// MultiHeadAttention_7937099563040
// MI455X (gfx1250) — matrix operand order repaired and checked by reading, not compiled, not run on hardware
//
#include <hip/hip_runtime.h>

// ---------------------------------------------------------------------------
// CDNA5 (gfx1250) multi-head attention, bf16 WMMA pipeline, bf16 workspace.
//   1) qkv GEMM (fp32 in, bf16 math) -> de-interleaved bf16 Q/K/V workspace
//        Q,K : [b,h,n,d] bf16      V : [b,h,d,n] bf16 (pre-transposed)
//   2) flash attention (bf16 WMMA, online softmax) -> attn [b*n, 2048] bf16
//   3) proj GEMM: A=attn bf16 (async-copied to LDS), B=w_proj fp32 -> f32 out
// Workspace: 3 * 16 MiB (Q/K/V) + 16 MiB (attn) = 64 MiB, fully L2-resident.
// ---------------------------------------------------------------------------

typedef __bf16 v16bf __attribute__((ext_vector_type(16)));
typedef float  v8f   __attribute__((ext_vector_type(8)));

union Frag16 {
    v16bf        v;
    unsigned int u[8];
};

#define GLOBAL_AS __attribute__((address_space(1)))
#define LDS_AS    __attribute__((address_space(3)))

// 16-byte payload type matching the async-load builtin's prototype:
// (int __vector(4) AS1*, int __vector(4) AS3*, Imm offset, Imm cpol)
typedef int b128_t __attribute__((vector_size(16)));

#if defined(__HIP_DEVICE_COMPILE__) && __has_builtin(__builtin_amdgcn_global_load_async_to_lds_b128)
#define ASYNC_CP 1
#endif

__device__ __forceinline__ void async_cp16(const unsigned short* gp, unsigned short* lp) {
#ifdef ASYNC_CP
    __builtin_amdgcn_global_load_async_to_lds_b128(
        (GLOBAL_AS b128_t*)gp, (LDS_AS b128_t*)lp, 0, 0);
#else
    *(uint4*)lp = *(const uint4*)gp;
#endif
}

__device__ __forceinline__ void wait_async0() {
#ifdef ASYNC_CP
#if __has_builtin(__builtin_amdgcn_s_wait_asynccnt)
    __builtin_amdgcn_s_wait_asynccnt(0);
#else
    asm volatile("s_wait_asynccnt 0" ::: "memory");
#endif
#endif
}

__device__ __forceinline__ unsigned short to_bf16(float a) {
    union { __bf16 h; unsigned short s; } p;
    p.h = (__bf16)a;
    return p.s;
}

__device__ __forceinline__ unsigned int pack_bf16(float a, float b) {
    union { __bf16 h[2]; unsigned int u; } p;
    p.h[0] = (__bf16)a;
    p.h[1] = (__bf16)b;
    return p.u;
}

// 16-bit A-matrix 16x32 layout (ISA 7.12.2): per-lane VGPR j holds packed K
// pair starting at: lanes 0-15 -> K = {0,2,4,6,16,18,20,22}; lanes 16-31 +8.
__device__ __forceinline__ int a_pair_k(int j, int half) {
    return (j < 4 ? 2 * j : 2 * j + 8) + (half ? 8 : 0);
}

__device__ __forceinline__ unsigned int lds_pair(const unsigned short* p) {
    return *(const unsigned int*)(const void*)p;
}

__device__ __forceinline__ v8f wmma_bf16(const Frag16& a, const Frag16& b, v8f c) {
    return __builtin_amdgcn_wmma_f32_16x16x32_bf16(
        false, a.v, false, b.v, (short)0, c, false, false);
}

#define GBM 128
#define GBN 128
#define GBK 32
#define LPAD 8      // LDS row pad (keeps 16B alignment: row stride 40 shorts)

// ---------------------------------------------------------------------------
// GEMM 1: qkv = x @ w_qkv + b_qkv, scatter-epilogue to bf16 Q/K/V tensors.
// Output column col = h*768 + d*3 + s  (s: 0=q, 1=k, 2=v).
// ---------------------------------------------------------------------------
__global__ __launch_bounds__(256)
void gemm_qkv_wmma(const float* __restrict__ A, const float* __restrict__ B,
                   const float* __restrict__ bias,
                   unsigned short* __restrict__ Qb, unsigned short* __restrict__ Kb,
                   unsigned short* __restrict__ Vb,
                   int M, int N, int K) {
    __shared__ unsigned short As[GBM][GBK + LPAD];   // [m][k]
    __shared__ unsigned short Bs[GBN][GBK + LPAD];   // [n][k] (B transposed)

    const int t = threadIdx.x, lane = t & 31, wave = t >> 5;
    const int half = lane >> 4, lm = lane & 15;
    const int waveM = wave & 3, waveN = wave >> 2;
    const int blockN = blockIdx.x * GBN, blockM = blockIdx.y * GBM;

    v8f acc[2][4];
#pragma unroll
    for (int mt = 0; mt < 2; ++mt)
#pragma unroll
        for (int nt = 0; nt < 4; ++nt)
#pragma unroll
            for (int e = 0; e < 8; ++e) acc[mt][nt][e] = 0.0f;

    for (int k0 = 0; k0 < K; k0 += GBK) {
        // Stage A tile as packed bf16 pairs (16 pairs per row)
        for (int p = t; p < GBM * (GBK / 2); p += 256) {
            int m = p >> 4, kp = p & 15;
            const float* ap = &A[(size_t)(blockM + m) * K + (k0 + 2 * kp)];
            *(unsigned int*)&As[m][2 * kp] = pack_bf16(ap[0], ap[1]);
        }
        // Stage B tile transposed as packed bf16 pairs
        for (int p = t; p < (GBK / 2) * GBN; p += 256) {
            int nn = p & 127, kp = p >> 7;
            float f0 = B[(size_t)(k0 + 2 * kp) * N + (blockN + nn)];
            float f1 = B[(size_t)(k0 + 2 * kp + 1) * N + (blockN + nn)];
            *(unsigned int*)&Bs[nn][2 * kp] = pack_bf16(f0, f1);
        }
        if (k0 + GBK < K) {
            __builtin_prefetch(&A[(size_t)(blockM + (t >> 1)) * K + (k0 + GBK)], 0, 0);
            __builtin_prefetch(&B[(size_t)(k0 + GBK + (t >> 7)) * N + blockN + (t & 127)], 0, 0);
        }
        __syncthreads();

        Frag16 af[2], bf[4];
#pragma unroll
        for (int mt = 0; mt < 2; ++mt) {
            int row = waveM * 32 + mt * 16 + lm;
#pragma unroll
            for (int j = 0; j < 8; ++j)
                af[mt].u[j] = lds_pair(&As[row][a_pair_k(j, half)]);
        }
#pragma unroll
        for (int nt = 0; nt < 4; ++nt) {
            int col = waveN * 64 + nt * 16 + lm;
#pragma unroll
            for (int j = 0; j < 8; ++j)
                bf[nt].u[j] = lds_pair(&Bs[col][a_pair_k(j, half)]);
        }
#pragma unroll
        for (int mt = 0; mt < 2; ++mt)
#pragma unroll
            for (int nt = 0; nt < 4; ++nt)
                acc[mt][nt] = wmma_bf16(af[mt], bf[nt], acc[mt][nt]);
        __syncthreads();
    }

    // Scatter epilogue: col -> (h, d, s); row -> (bi, nn). bf16 stores.
#pragma unroll
    for (int mt = 0; mt < 2; ++mt)
#pragma unroll
        for (int nt = 0; nt < 4; ++nt) {
            int col = blockN + waveN * 64 + nt * 16 + lm;
            int h = col / 768, rem = col - h * 768;
            int d = rem / 3, s = rem - d * 3;
            float bv = bias[col];
#pragma unroll
            for (int r = 0; r < 8; ++r) {
                int row = blockM + waveM * 32 + mt * 16 + r + half * 8;
                int bi = row >> 11, nn = row & 2047;
                int bh = bi * 8 + h;
                unsigned short bfv = to_bf16(acc[mt][nt][r] + bv);
                if (s == 0)       Qb[((size_t)bh * 2048 + nn) * 256 + d] = bfv;
                else if (s == 1)  Kb[((size_t)bh * 2048 + nn) * 256 + d] = bfv;
                else              Vb[((size_t)bh * 256 + d) * 2048 + nn] = bfv;
            }
        }
}

// ---------------------------------------------------------------------------
// Flash attention: Q,K [b,h,n,256] bf16; V [b,h,256,n] bf16; out bf16.
// Softmax on UNSCALED logits; divide by sqrt(EMB) afterwards (per reference).
// ---------------------------------------------------------------------------
#define HD   256
#define KBLK 32
#define NDT  16
#define NKC  8

__global__ __launch_bounds__(256)
void flash_attn_wmma(const unsigned short* __restrict__ Qb,
                     const unsigned short* __restrict__ Kb,
                     const unsigned short* __restrict__ Vb,
                     unsigned short* __restrict__ out, int n) {
    __shared__ unsigned short Ks[KBLK][HD + LPAD];     // [key][d], stride 528B
    __shared__ unsigned short Vt[HD][KBLK + LPAD];     // [d][key], stride 80B
    __shared__ unsigned short Ps[8][16][KBLK + 4];     // per-wave P tile

    const int t = threadIdx.x, lane = t & 31, wave = t >> 5;
    const int half = lane >> 4, lm = lane & 15;
    const int h = blockIdx.y, bi = blockIdx.z;
    const int bh = bi * 8 + h;
    const int qrow0 = blockIdx.x * 128 + wave * 16;

    // Q fragments: A-layout; K-pairs {0,2,4,6}+8h are 16 contiguous bytes.
    Frag16 qf[NKC];
    {
        const unsigned short* qrow = Qb + ((size_t)bh * n + qrow0 + lm) * HD;
#pragma unroll
        for (int c = 0; c < NKC; ++c)
#pragma unroll
            for (int j = 0; j < 8; ++j)
                qf[c].u[j] = *(const unsigned int*)&qrow[32 * c + a_pair_k(j, half)];
    }

    v8f acc[NDT];
#pragma unroll
    for (int nt = 0; nt < NDT; ++nt)
#pragma unroll
        for (int e = 0; e < 8; ++e) acc[nt][e] = 0.0f;
    float mrow[8], lrow[8];
#pragma unroll
    for (int r = 0; r < 8; ++r) { mrow[r] = -1e30f; lrow[r] = 0.0f; }

    for (int kb0 = 0; kb0 < n; kb0 += KBLK) {
        __syncthreads();
        // K tile: 32 rows x 512B  (async 16B chunks)
        {
            const unsigned short* Krow = Kb + ((size_t)bh * n + kb0) * HD;
            for (int c = t; c < 1024; c += 256) {
                int key = c >> 5, off = (c & 31) * 8;
                async_cp16(Krow + (size_t)key * HD + off, &Ks[key][off]);
            }
        }
        // V tile: 256 rows x 64B (already transposed in workspace)
        {
            const unsigned short* Vrow = Vb + (size_t)bh * HD * n + kb0;
            for (int c = t; c < 1024; c += 256) {
                int d = c >> 2, off = (c & 3) * 8;
                async_cp16(Vrow + (size_t)d * n + off, &Vt[d][off]);
            }
        }
        wait_async0();
        __syncthreads();

        // S = Q @ K^T : two 16x16 tiles (keys 0..15 / 16..31)
        v8f s0, s1;
#pragma unroll
        for (int e = 0; e < 8; ++e) { s0[e] = 0.0f; s1[e] = 0.0f; }
#pragma unroll
        for (int c = 0; c < NKC; ++c) {
            Frag16 k0f, k1f;
#pragma unroll
            for (int j = 0; j < 8; ++j) {
                int dd = 32 * c + a_pair_k(j, half);
                k0f.u[j] = lds_pair(&Ks[lm][dd]);
                k1f.u[j] = lds_pair(&Ks[16 + lm][dd]);
            }
            s0 = wmma_bf16(qf[c], k0f, s0);
            s1 = wmma_bf16(qf[c], k1f, s1);
        }

        // Online softmax; lane owns rows r + 8*half; xor masks stay in-half.
        float alpha[8];
#pragma unroll
        for (int r = 0; r < 8; ++r) {
            float v0 = s0[r], v1 = s1[r];
            float mx = fmaxf(v0, v1);
#pragma unroll
            for (int msk = 1; msk < 16; msk <<= 1)
                mx = fmaxf(mx, __shfl_xor(mx, msk, 32));
            float mnew = fmaxf(mrow[r], mx);
            float a  = __expf(mrow[r] - mnew);
            float p0 = __expf(v0 - mnew);
            float p1 = __expf(v1 - mnew);
            float ps = p0 + p1;
#pragma unroll
            for (int msk = 1; msk < 16; msk <<= 1)
                ps += __shfl_xor(ps, msk, 32);
            lrow[r]  = lrow[r] * a + ps;
            mrow[r]  = mnew;
            alpha[r] = a;
            int row = r + half * 8;
            Ps[wave][row][lm]      = to_bf16(p0);
            Ps[wave][row][16 + lm] = to_bf16(p1);
        }
        asm volatile("s_wait_dscnt 0" ::: "memory");   // wave-private LDS RAW

#pragma unroll
        for (int nt = 0; nt < NDT; ++nt)
#pragma unroll
            for (int r = 0; r < 8; ++r) acc[nt][r] *= alpha[r];

        Frag16 pf;
#pragma unroll
        for (int j = 0; j < 8; ++j)
            pf.u[j] = lds_pair(&Ps[wave][lm][a_pair_k(j, half)]);

#pragma unroll
        for (int nt = 0; nt < NDT; ++nt) {
            Frag16 vf;
            int dcol = nt * 16 + lm;
#pragma unroll
            for (int j = 0; j < 8; ++j)
                vf.u[j] = lds_pair(&Vt[dcol][a_pair_k(j, half)]);
            acc[nt] = wmma_bf16(pf, vf, acc[nt]);
        }
    }

    const float scaling = 45.254833995939045f;   // sqrt(2048)
#pragma unroll
    for (int r = 0; r < 8; ++r) {
        float inv = 1.0f / (lrow[r] * scaling);
        int row = qrow0 + r + half * 8;
#pragma unroll
        for (int nt = 0; nt < NDT; ++nt)
            out[((size_t)bi * n + row) * 2048 + h * HD + nt * 16 + lm] =
                to_bf16(acc[nt][r] * inv);
    }
}

// ---------------------------------------------------------------------------
// GEMM 2: C = A_bf16 @ B_f32 + bias, fp32 out. A tile async-copied to LDS.
// ---------------------------------------------------------------------------
__global__ __launch_bounds__(256)
void gemm_bf16xf32_wmma(const unsigned short* __restrict__ A,
                        const float* __restrict__ B, const float* __restrict__ bias,
                        float* __restrict__ C, int M, int N, int K) {
    __shared__ unsigned short As[GBM][GBK + LPAD];
    __shared__ unsigned short Bs[GBN][GBK + LPAD];

    const int t = threadIdx.x, lane = t & 31, wave = t >> 5;
    const int half = lane >> 4, lm = lane & 15;
    const int waveM = wave & 3, waveN = wave >> 2;
    const int blockN = blockIdx.x * GBN, blockM = blockIdx.y * GBM;

    v8f acc[2][4];
#pragma unroll
    for (int mt = 0; mt < 2; ++mt)
#pragma unroll
        for (int nt = 0; nt < 4; ++nt)
#pragma unroll
            for (int e = 0; e < 8; ++e) acc[mt][nt][e] = 0.0f;

    for (int k0 = 0; k0 < K; k0 += GBK) {
        // A tile: 128 rows x 64B, pure bf16 copy -> async 16B chunks
        for (int c = t; c < GBM * 4; c += 256) {
            int m = c >> 2, off = (c & 3) * 8;
            async_cp16(A + (size_t)(blockM + m) * K + k0 + off, &As[m][off]);
        }
        // B tile transposed, fp32 -> packed bf16 pairs
        for (int p = t; p < (GBK / 2) * GBN; p += 256) {
            int nn = p & 127, kp = p >> 7;
            float f0 = B[(size_t)(k0 + 2 * kp) * N + (blockN + nn)];
            float f1 = B[(size_t)(k0 + 2 * kp + 1) * N + (blockN + nn)];
            *(unsigned int*)&Bs[nn][2 * kp] = pack_bf16(f0, f1);
        }
        if (k0 + GBK < K)
            __builtin_prefetch(&B[(size_t)(k0 + GBK + (t >> 7)) * N + blockN + (t & 127)], 0, 0);
        wait_async0();
        __syncthreads();

        Frag16 af[2], bf[4];
#pragma unroll
        for (int mt = 0; mt < 2; ++mt) {
            int row = waveM * 32 + mt * 16 + lm;
#pragma unroll
            for (int j = 0; j < 8; ++j)
                af[mt].u[j] = lds_pair(&As[row][a_pair_k(j, half)]);
        }
#pragma unroll
        for (int nt = 0; nt < 4; ++nt) {
            int col = waveN * 64 + nt * 16 + lm;
#pragma unroll
            for (int j = 0; j < 8; ++j)
                bf[nt].u[j] = lds_pair(&Bs[col][a_pair_k(j, half)]);
        }
#pragma unroll
        for (int mt = 0; mt < 2; ++mt)
#pragma unroll
            for (int nt = 0; nt < 4; ++nt)
                acc[mt][nt] = wmma_bf16(af[mt], bf[nt], acc[mt][nt]);
        __syncthreads();
    }

#pragma unroll
    for (int mt = 0; mt < 2; ++mt)
#pragma unroll
        for (int nt = 0; nt < 4; ++nt) {
            int col = blockN + waveN * 64 + nt * 16 + lm;
            float bv = bias[col];
#pragma unroll
            for (int r = 0; r < 8; ++r) {
                int row = blockM + waveM * 32 + mt * 16 + r + half * 8;
                C[(size_t)row * N + col] = acc[mt][nt][r] + bv;
            }
        }
}

// ---------------------------------------------------------------------------
extern "C" void kernel_launch(void* const* d_in, const int* in_sizes, int n_in,
                              void* d_out, int out_size, void* d_ws, size_t ws_size,
                              hipStream_t stream) {
    const float* x      = (const float*)d_in[0];   // [2, 2048, 2048]
    const float* w_qkv  = (const float*)d_in[1];   // [2048, 6144]
    const float* b_qkv  = (const float*)d_in[2];   // [6144]
    const float* w_proj = (const float*)d_in[3];   // [2048, 2048]
    const float* b_proj = (const float*)d_in[4];   // [2048]
    float* out = (float*)d_out;                    // [2, 2048, 2048]

    const int M = 2 * 2048;                        // b * n
    const size_t headElems = (size_t)16 * 2048 * 256;   // 8M bf16 per tensor
    unsigned short* Qb   = (unsigned short*)d_ws;
    unsigned short* Kb   = Qb + headElems;
    unsigned short* Vb   = Kb + headElems;
    unsigned short* attn = Vb + headElems;         // [4096, 2048] bf16

    gemm_qkv_wmma<<<dim3(6144 / GBN, M / GBM), 256, 0, stream>>>(
        x, w_qkv, b_qkv, Qb, Kb, Vb, M, 6144, 2048);

    flash_attn_wmma<<<dim3(2048 / 128, 8, 2), 256, 0, stream>>>(
        Qb, Kb, Vb, attn, 2048);

    gemm_bf16xf32_wmma<<<dim3(2048 / GBN, M / GBM), 256, 0, stream>>>(
        attn, w_proj, b_proj, out, M, 2048, 2048);
}
